// CloudGenerator_10849087390080
// MI455X (gfx1250) — compile-verified, matrix operand order checked by reading, not run on hardware
//
#include <hip/hip_runtime.h>
#include <hip/hip_bf16.h>
#include <math.h>

typedef _Float16 v8h  __attribute__((ext_vector_type(8)));
typedef _Float16 v16h __attribute__((ext_vector_type(16)));
typedef float    v8f  __attribute__((ext_vector_type(8)));
typedef unsigned int v4ui __attribute__((ext_vector_type(4)));
typedef int      v4i  __attribute__((ext_vector_type(4)));
typedef int      v8i  __attribute__((ext_vector_type(8)));

#define N_NODES 65536
#define N_EDGES 524288
#define SQRT2F 1.41421356237309504880f

#if defined(__has_builtin)
# if __has_builtin(__builtin_amdgcn_tensor_load_to_lds)
#  define HAVE_TDM 1
# endif
#endif
#ifndef HAVE_TDM
# define HAVE_TDM 0
#endif

#if HAVE_TDM
// ---------------------------------------------------------------------------
// Tensor Data Mover: DMA a 16-row x 32-f32 tile (row stride ldX floats) from
// global memory into LDS at lds_byte_addr. D# bitfields per CDNA5 ISA ch.8:
//  group0: [1:0]=count=1, [63:32]=lds_addr, [120:64]=global_addr, [127:126]=2
//  group1: [17:16]=data_size(2=4B), [79:48]=tensor_dim0=32,
//          [111:80]=tensor_dim1=16, [127:112]=tile_dim0=32,
//          [143:128]=tile_dim1=16, [207:160]=tensor_dim0_stride=ldX
//  groups 2/3: zero (2D tensor). Completion tracked with TENSORcnt.
// ---------------------------------------------------------------------------
__device__ __forceinline__ void tdm_load_tile(const float* gptr,
                                              unsigned lds_byte_addr, int ldX)
{
  unsigned long long ga = (unsigned long long)(size_t)gptr;
  v4ui g0;
  g0[0] = 1u;                                            // count=1, user mode
  g0[1] = lds_byte_addr;                                 // lds_addr
  g0[2] = (unsigned)(ga & 0xFFFFFFFFu);                  // global_addr[31:0]
  g0[3] = (unsigned)((ga >> 32) & 0x01FFFFFFu) | (2u << 30);  // ga[56:32]|type=2
  v8i g1;
  g1[0] = (int)(2u << 16);      // workgroup_mask=0, data_size=2 (4 bytes)
  g1[1] = (int)(32u << 16);     // tensor_dim0[15:0] = 32
  g1[2] = (int)(16u << 16);     // tensor_dim0[31:16]=0, tensor_dim1[15:0]=16
  g1[3] = (int)(32u << 16);     // tensor_dim1[31:16]=0, tile_dim0=32
  g1[4] = 16;                   // tile_dim1=16, tile_dim2=0
  g1[5] = ldX;                  // tensor_dim0_stride[31:0] (element units)
  g1[6] = 0;                    // stride[47:32]=0, tensor_dim1_stride lo=0
  g1[7] = 0;
  v4i gz = {0, 0, 0, 0};
#if __clang_major__ >= 23
  v8i gz8 = {0, 0, 0, 0, 0, 0, 0, 0};
  __builtin_amdgcn_tensor_load_to_lds(g0, g1, gz, gz, gz8, 0);
#else
  __builtin_amdgcn_tensor_load_to_lds(g0, g1, gz, gz, 0);
#endif
}
#endif  // HAVE_TDM

// ---------------------------------------------------------------------------
// Weight prepack: write W (f32, [OUT,K] row-major) as the exact CDNA5 WMMA
// B-fragment register image, f16, zero padded to NT*16 cols x KT*32 K.
// n = j*16 + (lane&15), kk = kt*32 + (lane<16?0:8) + (h>>3)*16 + (h&7).
// One-time and tiny; all guards live here so the GEMM hot loop has none.
// ---------------------------------------------------------------------------
__global__ void pack_w(const float* __restrict__ W, _Float16* __restrict__ out,
                       int OUT, int K, int NT, int KT)
{
  int idx = blockIdx.x * blockDim.x + threadIdx.x;   // (j*KT+kt)*32 + lane
  int tot = NT * KT * 32;
  if (idx >= tot) return;
  int lane = idx & 31;
  int t = idx >> 5;
  int kt = t % KT;
  int j  = t / KT;
  int n  = j * 16 + (lane & 15);
  int kb = kt * 32 + ((lane < 16) ? 0 : 8);
  v8h lo, hi;
  #pragma unroll
  for (int h = 0; h < 8; ++h) {
    int kk = kb + h;
    lo[h] = (_Float16)((n < OUT && kk < K) ? W[(size_t)n * K + kk] : 0.f);
  }
  #pragma unroll
  for (int h = 0; h < 8; ++h) {
    int kk = kb + 16 + h;
    hi[h] = (_Float16)((n < OUT && kk < K) ? W[(size_t)n * K + kk] : 0.f);
  }
  *(v8h*)(out + (size_t)idx * 16)     = lo;
  *(v8h*)(out + (size_t)idx * 16 + 8) = hi;
}

// ---------------------------------------------------------------------------
// Fused ModLinear GEMM on WMMA f32_16x16x32_f16, double-buffered A staging.
// Y = act( demod( (X @ W^T * 1/sqrt(K) + lin_b) * style ) + ml_bias )
//     [ * psc + pof ]  [ + resid ]
// Block: 256 threads = 8 wave32; 16 rows x OUT (<=256) cols per block.
// Launcher guarantees: Nr multiple of 16; ldX >= KT*32 with rows readable
// (zero-padded) up to KT*32; Bpack is this layer's pack_w image.
// ---------------------------------------------------------------------------
__global__ __launch_bounds__(256) void gemm_ml(
    const float* __restrict__ X, int K, int ldX,
    const _Float16* __restrict__ Bpack,
    const float* __restrict__ linb,
    const float* __restrict__ style, const float* __restrict__ mlb,
    int OUT, int demod, int act,
    const float* __restrict__ resid,
    const float* __restrict__ psc, const float* __restrict__ pof,
    float* __restrict__ Y, int ldY)
{
  __shared__ float ldsA[2][16][32];     // 4 KB   double-buffered A tile (f32)
  __shared__ float ldsY[16][256];       // 16 KB  full output tile for demod
  __shared__ float red[16][16];
  __shared__ float rowscale[16];

  const int tid  = threadIdx.x;
  const int lane = tid & 31;
  const int wid  = tid >> 5;
  const int rowBase = blockIdx.x * 16;
  const int KT = (K + 31) >> 5;
  const int NT = (OUT + 15) >> 4;
  const bool has0 = wid < NT;
  const bool has1 = (wid + 8) < NT;
  const int n0 = wid * 16 + (lane & 15);
  const int n1 = (wid + 8) * 16 + (lane & 15);

  const float* __restrict__ Xtile = X + (size_t)rowBase * ldX;

#if !HAVE_TDM
  const int am_st = tid >> 4;           // fallback staging coords
  const int ak_st = (tid & 15) << 1;
#endif

  // ---- prologue: stage tile 0 into buffer 0 ------------------------------
#if HAVE_TDM
  if (wid == 0) {
    tdm_load_tile(Xtile, (unsigned)(size_t)&ldsA[0][0][0], ldX);
    __builtin_amdgcn_s_wait_tensorcnt(0);
  }
#else
  *(float2*)&ldsA[0][am_st][ak_st] =
      *(const float2*)(Xtile + (size_t)am_st * ldX + ak_st);
#endif
  __syncthreads();

  v8f acc0 = {0.f, 0.f, 0.f, 0.f, 0.f, 0.f, 0.f, 0.f};
  v8f acc1 = acc0;

  const int am  = lane & 15;
  const int akb = (lane < 16) ? 0 : 8;
  const _Float16* __restrict__ bp0 = Bpack + (((size_t)wid * KT) * 32 + lane) * 16;
  const _Float16* __restrict__ bp1 = Bpack + (((size_t)(wid + 8) * KT) * 32 + lane) * 16;

  for (int kt = 0; kt < KT; ++kt) {
    const int p = kt & 1;

    // Prefetch tile kt+1 into the other buffer (overlaps with WMMA below).
    if (kt + 1 < KT) {
#if HAVE_TDM
      if (wid == 0)
        tdm_load_tile(Xtile + (size_t)(kt + 1) * 32,
                      (unsigned)(size_t)&ldsA[p ^ 1][0][0], ldX);
#else
      *(float2*)&ldsA[p ^ 1][am_st][ak_st] =
          *(const float2*)(Xtile + (size_t)am_st * ldX + (size_t)(kt + 1) * 32 + ak_st);
#endif
    }

    // A fragment: CDNA5 16-bit 16x32 layout; two contiguous 8-float runs
    // per lane -> 4x ds_load_b128 + packed f16 converts.
    const float* r0 = &ldsA[p][am][akb];
    const float* r1 = &ldsA[p][am][akb + 16];
    v16h a;
    #pragma unroll
    for (int h = 0; h < 8; ++h) {
      a[h]     = (_Float16)r0[h];
      a[h + 8] = (_Float16)r1[h];
    }

    if (has0) {
      const v8h* q = (const v8h*)(bp0 + (size_t)kt * 512);
      v16h b = __builtin_shufflevector(q[0], q[1],
                                       0, 1, 2, 3, 4, 5, 6, 7,
                                       8, 9, 10, 11, 12, 13, 14, 15);
      acc0 = __builtin_amdgcn_wmma_f32_16x16x32_f16(false, a, false, b,
                                                    (short)0, acc0, false, false);
    }
    if (has1) {
      const v8h* q = (const v8h*)(bp1 + (size_t)kt * 512);
      v16h b = __builtin_shufflevector(q[0], q[1],
                                       0, 1, 2, 3, 4, 5, 6, 7,
                                       8, 9, 10, 11, 12, 13, 14, 15);
      acc1 = __builtin_amdgcn_wmma_f32_16x16x32_f16(false, a, false, b,
                                                    (short)0, acc1, false, false);
    }

#if HAVE_TDM
    if (wid == 0) __builtin_amdgcn_s_wait_tensorcnt(0);  // next tile landed
#endif
    __syncthreads();
  }

  // Spill accumulators to LDS with gain + lin bias + style.
  const float gain = rsqrtf((float)K);
  if (has0 && n0 < OUT) {
    float lb = linb[n0], st = style[n0];
    #pragma unroll
    for (int vr = 0; vr < 8; ++vr) {
      int m = vr + ((lane < 16) ? 0 : 8);   // C/D layout: vgpr vr -> M = vr (+8)
      ldsY[m][n0] = (acc0[vr] * gain + lb) * st;
    }
  }
  if (has1 && n1 < OUT) {
    float lb = linb[n1], st = style[n1];
    #pragma unroll
    for (int vr = 0; vr < 8; ++vr) {
      int m = vr + ((lane < 16) ? 0 : 8);
      ldsY[m][n1] = (acc1[vr] * gain + lb) * st;
    }
  }
  __syncthreads();

  // Row-wise L2 demodulation (F.normalize): 16 threads per row.
  if (demod) {
    int r = tid >> 4, c = tid & 15;
    float s = 0.f;
    for (int n = c; n < OUT; n += 16) { float v = ldsY[r][n]; s += v * v; }
    red[r][c] = s;
    __syncthreads();
    if (c == 0) {
      float t = 0.f;
      #pragma unroll
      for (int i = 0; i < 16; ++i) t += red[r][i];
      rowscale[r] = 1.f / fmaxf(sqrtf(t), 1e-12f);
    }
    __syncthreads();
  } else {
    if (tid < 16) rowscale[tid] = 1.f;
    __syncthreads();
  }

  // Epilogue: bias, activation, optional pos_scale/offset, optional residual.
  for (int idx = tid; idx < 16 * OUT; idx += 256) {
    int m = idx / OUT, n = idx - m * OUT;
    int row = rowBase + m;
    float v = ldsY[m][n] * rowscale[m] + mlb[n];
    if (act == 1)      v = (v > 0.f ? v : 0.2f * v) * SQRT2F;   // lrelu * sqrt(2)
    else if (act == 2) v = tanhf(v);
    if (psc)   v = v * psc[n] + pof[n];
    if (resid) v += resid[(size_t)row * ldY + n];
    Y[(size_t)row * ldY + n] = v;
  }
}

// ---------------------------------------------------------------------------
// Edge message + scatter-add: one wave per edge, lanes stride channels.
// agg[dst] += [pos[src]-pos[dst]+delta[dst], x[src]*edge_scale]
// ---------------------------------------------------------------------------
__global__ __launch_bounds__(256) void edge_scatter(
    const float* __restrict__ pos, const float* __restrict__ delta3,
    const float* __restrict__ x, const float* __restrict__ esc,
    const int* __restrict__ src, const int* __restrict__ dst,
    float* __restrict__ agg, int E, int C, int ldA)
{
  int wid = threadIdx.x >> 5, lane = threadIdx.x & 31;
  int e = blockIdx.x * 8 + wid;
  if (e >= E) return;
  int s = src[e], d = dst[e];
  size_t ab = (size_t)d * ldA;
  if (lane < 3) {
    float m = pos[s * 3 + lane] - pos[d * 3 + lane] + delta3[(size_t)d * 3 + lane];
    atomicAdd(&agg[ab + lane], m);
  }
  for (int c = lane; c < C; c += 32) {
    atomicAdd(&agg[ab + 3 + c], x[(size_t)s * C + c] * esc[c]);
  }
}

// ---------------------------------------------------------------------------
// Style: s = w @ affW^T * 1/sqrt(Z) + affb   (single row, OUT <= 256)
// ---------------------------------------------------------------------------
__global__ void style_kernel(const float* __restrict__ w,
                             const float* __restrict__ affW,
                             const float* __restrict__ affb,
                             float* __restrict__ out, int OUT, int Z)
{
  int o = blockIdx.x * blockDim.x + threadIdx.x;
  if (o >= OUT) return;
  float s = 0.f;
  for (int k = 0; k < Z; ++k) s += w[k] * affW[(size_t)o * Z + k];
  out[o] = s * rsqrtf((float)Z) + affb[o];
}

// Global max pool, pass 1: 128 blocks x 128 threads (thread = channel).
__global__ void pool_max1(const float* __restrict__ X, float* __restrict__ part, int N)
{
  int ch = threadIdx.x;
  int b = blockIdx.x;
  int per = N / 128;
  float m = -INFINITY;
  for (int r = b * per; r < (b + 1) * per; ++r)
    m = fmaxf(m, X[(size_t)r * 128 + ch]);
  part[b * 128 + ch] = m;
}

__global__ void pool_max2(const float* __restrict__ part, float* __restrict__ out)
{
  int t = threadIdx.x;
  float m = -INFINITY;
  for (int b = 0; b < 128; ++b) m = fmaxf(m, part[b * 128 + t]);
  out[t] = m;
}

// Tiny FC on a single 128-vector: y = lrelu(in @ W^T / sqrt(IND) + b) * sqrt(2)
__global__ void fc_vec_kernel(const float* __restrict__ in,
                              const float* __restrict__ W,
                              const float* __restrict__ b,
                              float* __restrict__ out, int IND, int OUTD)
{
  __shared__ float sin_[256];
  int t = threadIdx.x;
  if (t < IND) sin_[t] = in[t];
  __syncthreads();
  if (t < OUTD) {
    float s = 0.f;
    for (int k = 0; k < IND; ++k) s += sin_[k] * W[(size_t)t * IND + k];
    s = s * rsqrtf((float)IND) + b[t];
    s = (s > 0.f ? s : 0.2f * s) * SQRT2F;
    out[t] = s;
  }
}

// concat([x, tile(h)], -1) written to BOTH pre_feat and the feat working buffer.
__global__ void concat_kernel(const float* __restrict__ X, const float* __restrict__ h,
                              float* __restrict__ prefeat, float* __restrict__ xf, int N)
{
  size_t total = (size_t)N * 256;
  for (size_t idx = blockIdx.x * (size_t)blockDim.x + threadIdx.x; idx < total;
       idx += (size_t)gridDim.x * blockDim.x) {
    int c = (int)(idx & 255);
    float v = (c < 128) ? X[(idx >> 8) * 128 + c] : h[c - 128];
    prefeat[idx] = v;
    xf[idx] = v;
  }
}

__global__ void copy_kernel(const float* __restrict__ in, float* __restrict__ out, size_t n)
{
  for (size_t i = blockIdx.x * (size_t)blockDim.x + threadIdx.x; i < n;
       i += (size_t)gridDim.x * blockDim.x)
    out[i] = in[i];
}

__global__ void zero_kernel(float* __restrict__ out, size_t n)
{
  for (size_t i = blockIdx.x * (size_t)blockDim.x + threadIdx.x; i < n;
       i += (size_t)gridDim.x * blockDim.x)
    out[i] = 0.f;
}

// ---------------------------------------------------------------------------
extern "C" void kernel_launch(void* const* d_in, const int* in_sizes, int n_in,
                              void* d_out, int out_size, void* d_ws, size_t ws_size,
                              hipStream_t stream)
{
  (void)in_sizes; (void)n_in; (void)out_size; (void)ws_size;
  const int N = N_NODES, E = N_EDGES;
  const int LDA_POS = 160;    // roundup32(131)
  const int LDA_FEAT = 288;   // roundup32(259)

  auto P = [&](int i) { return (const float*)d_in[i]; };

  // Inputs (depth-first insertion-order flattening of setup_inputs()):
  //   0 pos [N,3], 1 x [N,128], 2 edge_index [2,E], 3 batch, 4 w [1,128]
  //   params leaves start at 5; each ml = {lin.W, lin.b, aff.W, aff.b, bias}
  //   pgnn = {h0(5), h1(5), g0(5), g1(5), edge_scale(1)} -> 21 leaves
  const float* posp = P(0);
  const float* xin  = P(1);
  const int*   srcp = (const int*)d_in[2];
  const int*   dstp = srcp + E;
  const float* wlat = P(4);
  const int B_POS0 = 5, B_POS1 = 26, B_FEAT0 = 47, B_FEAT1 = 68;
  const int GC0W = 89, GC0B = 90, GC1W = 91, GC1B = 92;
  const int B_T0 = 93, B_T1 = 98, B_T2 = 103, I_PSC = 108, I_POF = 109;

  // Workspace layout (floats).
  float* ws      = (float*)d_ws;
  float* BUF_X   = ws;                               // [N,128]   pos-stage x
  float* BUF_A   = BUF_X   + (size_t)N * 128;        // [N,288]   agg / tail1 out
  float* BUF_G0  = BUF_A   + (size_t)N * 288;        // [N,256]   delta / g0 / tail0
  float* BUF_D3  = BUF_G0  + (size_t)N * 256;        // [N,3]
  float* BUF_PRT = BUF_D3  + (size_t)N * 3;          // [128,128] pool partials
  float* BUF_PL  = BUF_PRT + 128 * 128;              // [128]
  float* BUF_H0  = BUF_PL  + 128;                    // [128]
  float* BUF_H1  = BUF_H0  + 128;                    // [128]
  float* BUF_ST  = BUF_H1  + 128;                    // 19 x 256 style vectors
  _Float16* PKB  = (_Float16*)(BUF_ST + (size_t)19 * 256);  // packed weights

  // Output regions: (new_pos [N,3], pre_feat [N,256], x [N,256])
  float* out_np = (float*)d_out;
  float* out_pf = out_np + (size_t)N * 3;
  float* out_xf = out_pf + (size_t)N * 256;

  auto ST = [&](int i) { return BUF_ST + (size_t)i * 256; };

  // --- 0. Layer table: {lin.W leaf, OUT, K} in style/pack slot order -------
  //   slots 0..15 : pos0{h0,h1,g0,g1} pos1{...} feat0{...} feat1{...}
  //   slots 16..18: tail0, tail1, tail2
  int Lw[19], Lout[19], Lk[19];
  {
    const int bases[4] = {B_POS0, B_POS1, B_FEAT0, B_FEAT1};
    for (int g = 0; g < 4; ++g) {
      int C = (g < 2) ? 128 : 256;
      int s = g * 4;
      Lw[s + 0] = bases[g] + 0;  Lout[s + 0] = C / 2; Lk[s + 0] = C;
      Lw[s + 1] = bases[g] + 5;  Lout[s + 1] = 3;     Lk[s + 1] = C / 2;
      Lw[s + 2] = bases[g] + 10; Lout[s + 2] = C;     Lk[s + 2] = C + 3;
      Lw[s + 3] = bases[g] + 15; Lout[s + 3] = C;     Lk[s + 3] = C;
    }
    Lw[16] = B_T0; Lout[16] = 128; Lk[16] = 256;
    Lw[17] = B_T1; Lout[17] = 64;  Lk[17] = 128;
    Lw[18] = B_T2; Lout[18] = 3;   Lk[18] = 64;
  }

  // --- 1. Pack all 19 weight matrices into WMMA B-fragment images ---------
  _Float16* PK[19];
  {
    size_t off = 0;
    for (int i = 0; i < 19; ++i) {
      int NT = (Lout[i] + 15) / 16, KT = (Lk[i] + 31) / 32;
      PK[i] = PKB + off;
      int tot = NT * KT * 32;
      pack_w<<<(tot + 255) / 256, 256, 0, stream>>>(P(Lw[i]), PK[i],
                                                    Lout[i], Lk[i], NT, KT);
      off += (size_t)tot * 16;
    }
  }

  // --- 2. Style vectors for all 19 ModLinear layers ------------------------
  for (int i = 0; i < 19; ++i)
    style_kernel<<<1, 256, 0, stream>>>(wlat, P(Lw[i] + 2), P(Lw[i] + 3),
                                        ST(i), Lout[i], 128);

  // --- helpers -------------------------------------------------------------
  auto ml = [&](const float* X, int ldx, int si,
                int demod, int act, const float* resid,
                const float* psc, const float* pof, float* Yp, int ldy) {
    gemm_ml<<<dim3(N / 16), 256, 0, stream>>>(X, Lk[si], ldx, PK[si],
                                              P(Lw[si] + 1), ST(si), P(Lw[si] + 4),
                                              Lout[si], demod, act,
                                              resid, psc, pof, Yp, ldy);
  };

  auto pgnn = [&](float* xb, int C, int base, int sti) {
    int H = C / 2, ldA = (C == 128) ? LDA_POS : LDA_FEAT;
    // delta head
    ml(xb, C, sti + 0, 1, 1, nullptr, nullptr, nullptr, BUF_G0, H);
    ml(BUF_G0, H, sti + 1, 0, 2, nullptr, nullptr, nullptr, BUF_D3, 3);
    // gather -> message -> scatter-add (agg rows zero-padded to ldA)
    zero_kernel<<<2048, 256, 0, stream>>>(BUF_A, (size_t)N * ldA);
    edge_scatter<<<E / 8, 256, 0, stream>>>(posp, BUF_D3, xb, P(base + 20),
                                            srcp, dstp, BUF_A, E, C, ldA);
    // output MLP + residual (in-place on xb)
    ml(BUF_A, ldA, sti + 2, 1, 1, nullptr, nullptr, nullptr, BUF_G0, C);
    ml(BUF_G0, C, sti + 3, 1, 1, xb, nullptr, nullptr, xb, C);
  };

  // --- 3. pos stage --------------------------------------------------------
  copy_kernel<<<2048, 256, 0, stream>>>(xin, BUF_X, (size_t)N * 128);
  pgnn(BUF_X, 128, B_POS0, 0);
  pgnn(BUF_X, 128, B_POS1, 4);

  // --- 4. global max pool + gc0/gc1 ----------------------------------------
  pool_max1<<<128, 128, 0, stream>>>(BUF_X, BUF_PRT, N);
  pool_max2<<<1, 128, 0, stream>>>(BUF_PRT, BUF_PL);
  fc_vec_kernel<<<1, 128, 0, stream>>>(BUF_PL, P(GC0W), P(GC0B), BUF_H0, 128, 128);
  fc_vec_kernel<<<1, 128, 0, stream>>>(BUF_H0, P(GC1W), P(GC1B), BUF_H1, 128, 128);

  // --- 5. concat -> pre_feat (output 2) and feat working buffer (output 3)
  concat_kernel<<<8192, 256, 0, stream>>>(BUF_X, BUF_H1, out_pf, out_xf, N);

  // --- 6. tail: new_pos ----------------------------------------------------
  ml(out_pf, 256, 16, 1, 1, nullptr, nullptr, nullptr, BUF_G0, 128);
  ml(BUF_G0, 128, 17, 1, 1, nullptr, nullptr, nullptr, BUF_A, 64);
  ml(BUF_A, 64, 18, 0, 2, nullptr, P(I_PSC), P(I_POF), out_np, 3);

  // --- 7. feat stage (in-place on out_xf, C = 256) -------------------------
  pgnn(out_xf, 256, B_FEAT0, 8);
  pgnn(out_xf, 256, B_FEAT1, 12);
}
